// WaveNet_55508157334074
// MI455X (gfx1250) — matrix operand order repaired and checked by reading, not compiled, not run on hardware
//
#include <hip/hip_runtime.h>
#include <math.h>

typedef __attribute__((ext_vector_type(16))) _Float16 v16h;
typedef __attribute__((ext_vector_type(8)))  float    v8f;

#define WMMA16(a, b, c) \
  __builtin_amdgcn_wmma_f32_16x16x32_f16(false, (a), false, (b), (short)0, (c), false, false)

namespace {
constexpr int T_FULL = 4096;
constexpr int DEC    = 256;   // decoder_seq_len
constexpr int TT     = 64;    // time tile per block
constexpr int HALO   = 128;   // max dilation
constexpr int TW     = TT + HALO;  // 192 staged timesteps
constexpr int CIN    = 32;
constexpr int COUT   = 64;
constexpr int CMID   = 64;
constexpr int PCH    = 128;
}

// A-fragment (16x32, f16) from a global f32 weight matrix.
// Rows base_m.., K columns base_k..base_k+31, row stride rstride, elem stride estride.
// CDNA5 layout: lane l<16 -> row l, K in {0..7, 16..23}; lane l+16 -> row l, K in {8..15, 24..31}.
__device__ __forceinline__ v16h loadA(const float* __restrict__ W, int rstride, int estride,
                                      int base_m, int base_k, int lane) {
  const int row = base_m + (lane & 15);
  const int kh  = (lane >> 4) * 8;
  const float* w0 = W + row * rstride + (base_k + kh) * estride;
  const float* w1 = W + row * rstride + (base_k + 16 + kh) * estride;
  v16h a;
#pragma unroll
  for (int j = 0; j < 8; ++j) a[j]     = (_Float16)w0[j * estride];
#pragma unroll
  for (int j = 0; j < 8; ++j) a[8 + j] = (_Float16)w1[j * estride];
  return a;
}

// B-fragment (32x16, f16) from LDS staged time-major [t][channel], row stride ldc.
// Lane l<16 -> column t_base+l, K = base_k+{0..7, 16..23} ; lane l+16 -> same column set, K = base_k+{8..15, 24..31}.
__device__ __forceinline__ v16h loadB(const _Float16* S, int ldc, int t_base, int base_k, int lane) {
  const _Float16* p = S + (t_base + (lane & 15)) * ldc + base_k + (lane >> 4) * 8;
  v16h b;
#pragma unroll
  for (int j = 0; j < 8; ++j) { b[j] = p[j]; b[8 + j] = p[16 + j]; }
  return b;
}

__global__ __launch_bounds__(128) void wavenet_tail_wmma(
    const float* __restrict__ x,
    const float* __restrict__ pre_w,  const float* __restrict__ pre_b,
    const float* __restrict__ filt_w, const float* __restrict__ filt_b,
    const float* __restrict__ gate_w, const float* __restrict__ gate_b,
    const float* __restrict__ post_w, const float* __restrict__ post_b,
    const float* __restrict__ d1_w,   const float* __restrict__ d1_b,
    const float* __restrict__ d2_w,   const float* __restrict__ d2_b,
    float* __restrict__ out) {
  __shared__ _Float16 xT[TW][CIN];    // 12 KB  x window, time-major
  __shared__ _Float16 hT[TW][COUT];   // 24 KB  branch 'pre' output
  __shared__ _Float16 sT[TT][CMID];   //  8 KB  gated activation
  __shared__ _Float16 aT[TT][COUT];   //  8 KB  skip accumulator (for d1)
  __shared__ _Float16 rT[TT][PCH];    // 16 KB  d1 output

  const int tid  = threadIdx.x;
  const int lane = tid & 31;
  const int wid  = tid >> 5;          // 4 waves
  const int ln   = lane & 15;
  const int lh   = lane >> 4;

  const int b    = blockIdx.x >> 2;        // batch 0..31
  const int tile = blockIdx.x & 3;         // time tile 0..3
  const int t0   = T_FULL - DEC + tile * TT;
  const int tg0  = t0 - HALO;              // >= 3712, tg0+TW <= 4096

  // ---- stage x window, time-major, f32 -> f16 ----
  for (int idx = tid; idx < TW * CIN; idx += 128) {
    const int lt = idx >> 5;     // CIN == 32
    const int c  = idx & 31;
    xT[lt][c] = (_Float16)x[(b * CIN + c) * T_FULL + tg0 + lt];
  }
  __syncthreads();

  const int dils[8] = {1, 2, 4, 8, 16, 32, 64, 128};
  const int m0 = wid * 16;      // this wave's output-channel tile for M=64 GEMMs

  float acc[4][8];              // skip-sum, 4 N-tiles x 8 rows, f32 in registers
#pragma unroll
  for (int nt = 0; nt < 4; ++nt)
#pragma unroll
    for (int j = 0; j < 8; ++j) acc[nt][j] = 0.f;

#pragma unroll 1
  for (int i = 0; i < 8; ++i) {
    const int d  = dils[i];
    const int hq = (d + 15) >> 4;          // halo in 16-step tiles
    const int lt0 = HALO - hq * 16;        // first staged timestep this branch needs

    // ---- pre: h = relu(pre_w @ x + pre_b), M=64 K=32, over TT + halo ----
    {
      const v16h aPre = loadA(pre_w + i * COUT * CIN, CIN, 1, m0, 0, lane);
      float bias[8];
#pragma unroll
      for (int j = 0; j < 8; ++j) bias[j] = pre_b[i * COUT + m0 + lh * 8 + j];
      const int ntiles = 4 + hq;
      for (int nt = 0; nt < ntiles; ++nt) {
        const int lt = lt0 + nt * 16;
        v8f c = {};
        c = WMMA16(aPre, loadB(&xT[0][0], CIN, lt, 0, lane), c);
        const int t = lt + ln;
#pragma unroll
        for (int j = 0; j < 8; ++j)
          hT[t][m0 + lh * 8 + j] = (_Float16)fmaxf(c[j] + bias[j], 0.f);
      }
    }
    __syncthreads();

    // ---- filt/gate: K=2 dilated conv == two shifted 1x1s; s = tanh(f)*relu(g) ----
    {
      v16h af[2][2], ag[2][2];   // [tap][kstep]
#pragma unroll
      for (int tap = 0; tap < 2; ++tap)
#pragma unroll
        for (int ks = 0; ks < 2; ++ks) {
          af[tap][ks] = loadA(filt_w + i * CMID * COUT * 2 + tap, COUT * 2, 2, m0, ks * 32, lane);
          ag[tap][ks] = loadA(gate_w + i * CMID * COUT * 2 + tap, COUT * 2, 2, m0, ks * 32, lane);
        }
      float fb[8], gb[8];
#pragma unroll
      for (int j = 0; j < 8; ++j) {
        fb[j] = filt_b[i * CMID + m0 + lh * 8 + j];
        gb[j] = gate_b[i * CMID + m0 + lh * 8 + j];
      }
#pragma unroll
      for (int nt = 0; nt < 4; ++nt) {
        const int ltc = HALO + nt * 16;   // current tap (w[...,1])
        const int ltd = ltc - d;          // delayed tap (w[...,0])
        v8f fc = {}, gc = {};
#pragma unroll
        for (int ks = 0; ks < 2; ++ks) {
          const v16h bd = loadB(&hT[0][0], COUT, ltd, ks * 32, lane);
          fc = WMMA16(af[0][ks], bd, fc);
          gc = WMMA16(ag[0][ks], bd, gc);
          const v16h bc = loadB(&hT[0][0], COUT, ltc, ks * 32, lane);
          fc = WMMA16(af[1][ks], bc, fc);
          gc = WMMA16(ag[1][ks], bc, gc);
        }
        const int t = nt * 16 + ln;
#pragma unroll
        for (int j = 0; j < 8; ++j) {
          const float fv = tanhf(fc[j] + fb[j]);
          const float gv = fmaxf(gc[j] + gb[j], 0.f);
          sT[t][m0 + lh * 8 + j] = (_Float16)(fv * gv);
        }
      }
    }
    __syncthreads();

    // ---- post: acc += relu(post_w @ s + post_b), M=64 K=64 ----
    {
      v16h ap[2];
#pragma unroll
      for (int ks = 0; ks < 2; ++ks)
        ap[ks] = loadA(post_w + i * COUT * CMID, CMID, 1, m0, ks * 32, lane);
      float pb[8];
#pragma unroll
      for (int j = 0; j < 8; ++j) pb[j] = post_b[i * COUT + m0 + lh * 8 + j];
#pragma unroll
      for (int nt = 0; nt < 4; ++nt) {
        v8f pc = {};
#pragma unroll
        for (int ks = 0; ks < 2; ++ks)
          pc = WMMA16(ap[ks], loadB(&sT[0][0], CMID, nt * 16, ks * 32, lane), pc);
#pragma unroll
        for (int j = 0; j < 8; ++j) acc[nt][j] += fmaxf(pc[j] + pb[j], 0.f);
      }
    }
    // next branch's pre-write of hT is fenced by the barrier after it
  }

  // ---- spill skip-sum to LDS for d1 ----
#pragma unroll
  for (int nt = 0; nt < 4; ++nt) {
    const int t = nt * 16 + ln;
#pragma unroll
    for (int j = 0; j < 8; ++j) aT[t][m0 + lh * 8 + j] = (_Float16)acc[nt][j];
  }
  __syncthreads();

  // ---- d1: r = relu(d1_w @ acc + d1_b), M=128 K=64 (each wave: 2 M-tiles) ----
#pragma unroll
  for (int mt = 0; mt < 2; ++mt) {
    const int m1 = wid * 32 + mt * 16;
    v16h ad[2];
#pragma unroll
    for (int ks = 0; ks < 2; ++ks) ad[ks] = loadA(d1_w, COUT, 1, m1, ks * 32, lane);
    float db[8];
#pragma unroll
    for (int j = 0; j < 8; ++j) db[j] = d1_b[m1 + lh * 8 + j];
#pragma unroll
    for (int nt = 0; nt < 4; ++nt) {
      v8f rc = {};
#pragma unroll
      for (int ks = 0; ks < 2; ++ks)
        rc = WMMA16(ad[ks], loadB(&aT[0][0], COUT, nt * 16, ks * 32, lane), rc);
      const int t = nt * 16 + ln;
#pragma unroll
      for (int j = 0; j < 8; ++j)
        rT[t][m1 + lh * 8 + j] = (_Float16)fmaxf(rc[j] + db[j], 0.f);
    }
  }
  __syncthreads();

  // ---- d2: M=1 -> plain 128-MAC dot per timestep, then relu ----
  if (tid < TT) {
    float s = d2_b[0];
#pragma unroll 8
    for (int p = 0; p < PCH; ++p) s += d2_w[p] * (float)rT[tid][p];
    out[b * DEC + tile * TT + tid] = fmaxf(s, 0.f);
  }
}

extern "C" void kernel_launch(void* const* d_in, const int* in_sizes, int n_in,
                              void* d_out, int out_size, void* d_ws, size_t ws_size,
                              hipStream_t stream) {
  (void)in_sizes; (void)n_in; (void)d_ws; (void)ws_size; (void)out_size;
  const float* x      = (const float*)d_in[0];
  const float* pre_w  = (const float*)d_in[1];
  const float* pre_b  = (const float*)d_in[2];
  const float* filt_w = (const float*)d_in[3];
  const float* filt_b = (const float*)d_in[4];
  const float* gate_w = (const float*)d_in[5];
  const float* gate_b = (const float*)d_in[6];
  const float* post_w = (const float*)d_in[7];
  const float* post_b = (const float*)d_in[8];
  const float* d1_w   = (const float*)d_in[9];
  const float* d1_b   = (const float*)d_in[10];
  const float* d2_w   = (const float*)d_in[11];
  const float* d2_b   = (const float*)d_in[12];
  // d_in[13] = teacher_forcing_ratio (== 1 path), unused

  float* out = (float*)d_out;
  dim3 grid(32 * (DEC / TT));   // 128 blocks: batch x time-tile
  dim3 block(128);              // 4 waves (wave32)
  hipLaunchKernelGGL(wavenet_tail_wmma, grid, block, 0, stream,
                     x, pre_w, pre_b, filt_w, filt_b, gate_w, gate_b,
                     post_w, post_b, d1_w, d1_b, d2_w, d2_b, out);
}
